// StructureEnhancementBlock_34978213659230
// MI455X (gfx1250) — matrix operand order repaired and checked by reading, not compiled, not run on hardware
//
#include <hip/hip_runtime.h>
#include <hip/hip_bf16.h>

// ---------------- problem constants ----------------
#define B_    8
#define C_    256
#define NH_   4
#define HD_   64
#define HW_   4096
#define EPS_LN   1e-5f
#define EPS_NORM 1e-12f

typedef __attribute__((ext_vector_type(16))) __bf16 v16bf;
typedef __attribute__((ext_vector_type(8)))  float  v8f;

union FragBF { v16bf v; uint4 u[2]; };
union AccF   { v8f   v; float f[8]; };

__device__ __forceinline__ unsigned short f2bf(float f) {
  unsigned u = __float_as_uint(f);
  u += 0x7fffu + ((u >> 16) & 1u);          // round-to-nearest-even
  return (unsigned short)(u >> 16);
}
__device__ __forceinline__ float bf2f(unsigned short h) {
  return __uint_as_float(((unsigned)h) << 16);
}
__device__ __forceinline__ float gelu_tanh(float x) {
  float x3 = x * x * x;
  return 0.5f * x * (1.0f + tanhf(0.7978845608028654f * (x + 0.044715f * x3)));
}
__device__ __forceinline__ void zero_acc(AccF& a) {
#pragma unroll
  for (int i = 0; i < 8; ++i) a.f[i] = 0.0f;
}

// ---------------- fp32 -> bf16 convert (weights) ----------------
__global__ __launch_bounds__(256)
void cvt_f32_bf16_kernel(const float* __restrict__ src, unsigned short* __restrict__ dst, int n) {
  int i = blockIdx.x * 256 + threadIdx.x;
  if (i < n) dst[i] = f2bf(src[i]);
}

// ---------------- LayerNorm over channel dim, bf16 out ----------------
__global__ __launch_bounds__(256)
void ln2d_kernel(const float* __restrict__ x, const float* __restrict__ w,
                 const float* __restrict__ bia, unsigned short* __restrict__ out) {
  int pos = blockIdx.x * 256 + threadIdx.x;        // over nbatch*HW_
  int b = pos / HW_, n = pos % HW_;
  const float* xp = x + (size_t)b * C_ * HW_ + n;
  float s1 = 0.f, s2 = 0.f;
  for (int c = 0; c < C_; ++c) { float v = xp[(size_t)c * HW_]; s1 += v; s2 += v * v; }
  float mean = s1 * (1.0f / C_);
  float var  = s2 * (1.0f / C_) - mean * mean;
  float rstd = rsqrtf(var + EPS_LN);
  unsigned short* op = out + (size_t)b * C_ * HW_ + n;
  for (int c = 0; c < C_; ++c) {
    float v = xp[(size_t)c * HW_];
    op[(size_t)c * HW_] = f2bf((v - mean) * rstd * w[c] + bia[c]);
  }
}

// ---------------- row L2 norms over bf16 rows of length HW_ ----------------
__global__ __launch_bounds__(256)
void rownorm_kernel(const unsigned short* __restrict__ src, float* __restrict__ norms,
                    int rows_per_batch, size_t batch_stride) {
  int row = blockIdx.x;
  int b = row / rows_per_batch, r = row % rows_per_batch;
  const unsigned short* p = src + (size_t)b * batch_stride + (size_t)r * HW_;
  float s = 0.f;
  for (int i = threadIdx.x; i < HW_; i += 256) { float v = bf2f(p[i]); s += v * v; }
  __shared__ float red[256];
  red[threadIdx.x] = s; __syncthreads();
  for (int st = 128; st > 0; st >>= 1) {
    if (threadIdx.x < st) red[threadIdx.x] += red[threadIdx.x + st];
    __syncthreads();
  }
  if (threadIdx.x == 0) norms[row] = fmaxf(sqrtf(red[0]), EPS_NORM);
}

// ---------------- batched GEMM: out[b] = W(MxK) x act[b](KxN), WMMA bf16 ----------------
// block tile 128(M) x 128(N), 8 waves in 4x2, wave tile 32x64 (2x4 WMMA tiles),
// K step 32, double-buffered LDS so global loads for step k+1 overlap WMMAs of step k.
template<bool BIAS, bool RES, bool GELU, bool OUTF32, bool OUTBF16>
__global__ __launch_bounds__(256)
void gemm_bf16_kernel(const unsigned short* __restrict__ W,
                      const unsigned short* __restrict__ act,
                      float* __restrict__ outF, unsigned short* __restrict__ outH,
                      const float* __restrict__ bias, const float* __restrict__ res,
                      int M, int K, int N) {
  const int nb = blockIdx.x, mb = blockIdx.y, bt = blockIdx.z;
  const int tid = threadIdx.x;
  const int wave = tid >> 5, lane = tid & 31;
  const int half = lane >> 4, l16 = lane & 15;
  const int wm = wave >> 1, wn = wave & 1;
  const int m0 = mb * 128, n0 = nb * 128;
  const unsigned short* actb = act + (size_t)bt * K * N;

  __shared__ __align__(16) unsigned short As[2][128][32];  // [m][k]
  __shared__ __align__(16) unsigned short Bs[2][128][32];  // [n][k] (transposed)

  AccF acc[2][4];
#pragma unroll
  for (int i = 0; i < 2; ++i)
#pragma unroll
    for (int j = 0; j < 4; ++j) zero_acc(acc[i][j]);

  // per-thread staging coordinates (A: 512 uint4, B: 512 uint4, 2 each per thread)
  int aRow[2], aCol[2], bRow[2], bCol[2];
#pragma unroll
  for (int i = 0; i < 2; ++i) {
    int idx = tid + i * 256;
    aRow[i] = idx >> 2;  aCol[i] = (idx & 3) << 3;   // A: 128 rows x 4 uint4
    bRow[i] = idx >> 4;  bCol[i] = (idx & 15) << 3;  // B: 32 k-rows x 16 uint4
  }

  uint4 ra[2], rb[2];
  auto gload = [&](int k0) {
#pragma unroll
    for (int i = 0; i < 2; ++i) {
      ra[i] = *(const uint4*)&W[(size_t)(m0 + aRow[i]) * K + k0 + aCol[i]];
      rb[i] = *(const uint4*)&actb[(size_t)(k0 + bRow[i]) * N + n0 + bCol[i]];
    }
  };

  const int nk = K >> 5;
  gload(0);
  for (int kk = 0; kk < nk; ++kk) {
    const int buf = kk & 1;
    // stage regs -> LDS[buf]
#pragma unroll
    for (int i = 0; i < 2; ++i) {
      *(uint4*)&As[buf][aRow[i]][aCol[i]] = ra[i];
      const unsigned short* pp = (const unsigned short*)&rb[i];
#pragma unroll
      for (int e = 0; e < 8; ++e) Bs[buf][bCol[i] + e][bRow[i]] = pp[e];
    }
    if (kk + 1 < nk) gload((kk + 1) << 5);   // prefetch next K-slab (overlaps WMMAs)
    __syncthreads();

    FragBF a[2], bfr[4];
#pragma unroll
    for (int i = 0; i < 2; ++i) {
      int m = wm * 32 + i * 16 + l16;
      a[i].u[0] = *(const uint4*)&As[buf][m][half * 8];        // K = 8h..8h+7
      a[i].u[1] = *(const uint4*)&As[buf][m][half * 8 + 16];   // K = 8h+16..8h+23
    }
#pragma unroll
    for (int j = 0; j < 4; ++j) {
      int n = wn * 64 + j * 16 + l16;
      bfr[j].u[0] = *(const uint4*)&Bs[buf][n][half * 8];      // K = 8h..8h+7
      bfr[j].u[1] = *(const uint4*)&Bs[buf][n][half * 8 + 16]; // K = 8h+16..8h+23
    }
#pragma unroll
    for (int i = 0; i < 2; ++i)
#pragma unroll
      for (int j = 0; j < 4; ++j)
        acc[i][j].v = __builtin_amdgcn_wmma_f32_16x16x32_bf16(
            false, a[i].v, false, bfr[j].v, (short)0, acc[i][j].v, false, false);
  }

  // epilogue: bias -> gelu -> residual -> store
#pragma unroll
  for (int i = 0; i < 2; ++i)
#pragma unroll
    for (int j = 0; j < 4; ++j) {
      int mBase = m0 + wm * 32 + i * 16 + 8 * half;
      int n = n0 + wn * 64 + j * 16 + l16;
#pragma unroll
      for (int r = 0; r < 8; ++r) {
        int m = mBase + r;
        float v = acc[i][j].f[r];
        if (BIAS) v += bias[m];
        if (GELU) v = gelu_tanh(v);
        size_t o = (size_t)bt * M * N + (size_t)m * N + n;
        if (RES) v += res[o];
        if (OUTF32) outF[o] = v;
        if (OUTBF16) outH[o] = f2bf(v);
      }
    }
}

// ---------------- attention logits + softmax ----------------
// grid (2, NH_, B_), block 128 (4 waves). S = (Q . K^T)/(|q||k|), softmax rows, bf16 out.
// Fragment registers are double-buffered so global loads run ahead of the WMMAs.
__global__ __launch_bounds__(128)
void attn_logits_kernel(const unsigned short* __restrict__ qh,   // B_ x 256 x HW_
                        const unsigned short* __restrict__ kvh,  // 16 x 512 x HW_
                        const float* __restrict__ qnorm,         // B_*256
                        const float* __restrict__ knorm,         // 16*256
                        unsigned short* __restrict__ attn) {     // B_,NH_,2,64,64
  const int t = blockIdx.x, nh = blockIdx.y, b = blockIdx.z;
  const int kb = t * B_ + b;
  const int tid = threadIdx.x, wave = tid >> 5, lane = tid & 31;
  const int half = lane >> 4, l16 = lane & 15;
  const unsigned short* qbase = qh + ((size_t)b * C_ + nh * HD_) * HW_;
  const unsigned short* kbase = kvh + ((size_t)kb * 2 * C_ + nh * HD_) * HW_;

  AccF acc[4];
#pragma unroll
  for (int j = 0; j < 4; ++j) zero_acc(acc[j]);

  const unsigned short* aRow = qbase + (size_t)(wave * 16 + l16) * HW_;
  FragBF a[2], bfr[2][4];
  auto loadf = [&](int k0, int s) {
    a[s].u[0] = *(const uint4*)&aRow[k0 + half * 8];
    a[s].u[1] = *(const uint4*)&aRow[k0 + half * 8 + 16];
#pragma unroll
    for (int j = 0; j < 4; ++j) {
      const unsigned short* bRow = kbase + (size_t)(j * 16 + l16) * HW_;
      bfr[s][j].u[0] = *(const uint4*)&bRow[k0 + half * 8];
      bfr[s][j].u[1] = *(const uint4*)&bRow[k0 + half * 8 + 16];
    }
  };

  loadf(0, 0);
  for (int k0 = 0; k0 < HW_; k0 += 32) {
    const int cur = (k0 >> 5) & 1;
    if (k0 + 32 < HW_) loadf(k0 + 32, cur ^ 1);   // prefetch next fragments
#pragma unroll
    for (int j = 0; j < 4; ++j)
      acc[j].v = __builtin_amdgcn_wmma_f32_16x16x32_bf16(
          false, a[cur].v, false, bfr[cur][j].v, (short)0, acc[j].v, false, false);
  }

  __shared__ float S[64][64];
  const float* qn = qnorm + b * C_ + nh * HD_;
  const float* kn = knorm + kb * C_ + nh * HD_;
#pragma unroll
  for (int j = 0; j < 4; ++j)
#pragma unroll
    for (int r = 0; r < 8; ++r) {
      int m = wave * 16 + 8 * half + r;
      int d = j * 16 + l16;
      S[m][d] = acc[j].f[r] / (qn[m] * kn[d]);
    }
  __syncthreads();

  if (tid < 64) {
    float mx = -3.0e38f;
    for (int d = 0; d < 64; ++d) mx = fmaxf(mx, S[tid][d]);
    float sum = 0.f;
    for (int d = 0; d < 64; ++d) { float e = __expf(S[tid][d] - mx); S[tid][d] = e; sum += e; }
    float inv = 1.0f / sum;
    unsigned short* ap = attn + ((((size_t)b * NH_ + nh) * 2 + t) * 64 + tid) * 64;
    for (int d = 0; d < 64; ++d) ap[d] = f2bf(S[tid][d] * inv);
  }
}

// ---------------- AV + sigmoid gate combine ----------------
// grid (HW_/64, NH_, B_), block 128. out[c,h] = g*At.Vt + (1-g)*Ad.Vd -> bf16 qkvh
__global__ __launch_bounds__(128)
void attn_av_kernel(const unsigned short* __restrict__ attn,   // B_,NH_,2,64,64
                    const unsigned short* __restrict__ kvh,    // 16 x 512 x HW_
                    const float* __restrict__ attn_scale,      // NH_
                    unsigned short* __restrict__ qkvh) {       // B_ x 256 x HW_
  const int nc = blockIdx.x, nh = blockIdx.y, b = blockIdx.z;
  const int n0 = nc * 64;
  const int tid = threadIdx.x, wave = tid >> 5, lane = tid & 31;
  const int half = lane >> 4, l16 = lane & 15;

  __shared__ __align__(16) unsigned short Vt[64][64];  // [n][d]
  __shared__ __align__(16) unsigned short Vd[64][64];
  const unsigned short* vtb = kvh + ((size_t)b * 2 * C_ + C_ + nh * HD_) * HW_;
  const unsigned short* vdb = kvh + ((size_t)(B_ + b) * 2 * C_ + C_ + nh * HD_) * HW_;
#pragma unroll
  for (int i = 0; i < 4; ++i) {
    int idx = tid + i * 128;           // 512 uint4s
    int d = idx >> 3;
    int c8 = (idx & 7) << 3;
    uint4 pk = *(const uint4*)&vtb[(size_t)d * HW_ + n0 + c8];
    const unsigned short* pp = (const unsigned short*)&pk;
#pragma unroll
    for (int e = 0; e < 8; ++e) Vt[c8 + e][d] = pp[e];
    pk = *(const uint4*)&vdb[(size_t)d * HW_ + n0 + c8];
#pragma unroll
    for (int e = 0; e < 8; ++e) Vd[c8 + e][d] = pp[e];
  }
  __syncthreads();

  const unsigned short* at = attn + (((size_t)b * NH_ + nh) * 2 + 0) * 4096;
  const unsigned short* ad = at + 4096;
  AccF accT[4], accD[4];
#pragma unroll
  for (int j = 0; j < 4; ++j) { zero_acc(accT[j]); zero_acc(accD[j]); }

  const int m = wave * 16 + l16;
  for (int k0 = 0; k0 < 64; k0 += 32) {
    FragBF aT, aD;
    aT.u[0] = *(const uint4*)&at[m * 64 + k0 + half * 8];
    aT.u[1] = *(const uint4*)&at[m * 64 + k0 + half * 8 + 16];
    aD.u[0] = *(const uint4*)&ad[m * 64 + k0 + half * 8];
    aD.u[1] = *(const uint4*)&ad[m * 64 + k0 + half * 8 + 16];
#pragma unroll
    for (int j = 0; j < 4; ++j) {
      int n = j * 16 + l16;
      FragBF bT, bD;
      bT.u[0] = *(const uint4*)&Vt[n][k0 + half * 8];
      bT.u[1] = *(const uint4*)&Vt[n][k0 + half * 8 + 16];
      bD.u[0] = *(const uint4*)&Vd[n][k0 + half * 8];
      bD.u[1] = *(const uint4*)&Vd[n][k0 + half * 8 + 16];
      accT[j].v = __builtin_amdgcn_wmma_f32_16x16x32_bf16(
          false, aT.v, false, bT.v, (short)0, accT[j].v, false, false);
      accD[j].v = __builtin_amdgcn_wmma_f32_16x16x32_bf16(
          false, aD.v, false, bD.v, (short)0, accD[j].v, false, false);
    }
  }

  float g = 1.0f / (1.0f + __expf(-attn_scale[nh]));
#pragma unroll
  for (int j = 0; j < 4; ++j)
#pragma unroll
    for (int r = 0; r < 8; ++r) {
      int mo = wave * 16 + 8 * half + r;
      int n = n0 + j * 16 + l16;
      float v = g * accT[j].f[r] + (1.0f - g) * accD[j].f[r];
      qkvh[((size_t)b * C_ + nh * HD_ + mo) * HW_ + n] = f2bf(v);
    }
}

// ---------------- host-side launch ----------------
extern "C" void kernel_launch(void* const* d_in, const int* in_sizes, int n_in,
                              void* d_out, int out_size, void* d_ws, size_t ws_size,
                              hipStream_t stream) {
  const float* img    = (const float*)d_in[0];
  const float* aux0   = (const float*)d_in[1];
  const float* aux1   = (const float*)d_in[2];
  const float* q_ln_w = (const float*)d_in[3];
  const float* q_ln_b = (const float*)d_in[4];
  const float* kv_ln_w= (const float*)d_in[5];
  const float* kv_ln_b= (const float*)d_in[6];
  const float* Wq     = (const float*)d_in[7];
  const float* Wkv    = (const float*)d_in[8];
  const float* Wo     = (const float*)d_in[9];
  const float* bo     = (const float*)d_in[10];
  const float* ascale = (const float*)d_in[11];
  const float* mlp_ln_w = (const float*)d_in[12];
  const float* mlp_ln_b = (const float*)d_in[13];
  const float* W1     = (const float*)d_in[14];
  const float* b1     = (const float*)d_in[15];
  const float* W2     = (const float*)d_in[16];
  const float* b2     = (const float*)d_in[17];
  float* out = (float*)d_out;

  // bump allocator in workspace
  char* base = (char*)d_ws;
  size_t off = 0;
  auto alloc = [&](size_t bytes) -> char* {
    char* p = base + off;
    off += (bytes + 255) & ~(size_t)255;
    return p;
  };
  const size_t actE = (size_t)B_ * C_ * HW_;             // 8.39M elems
  unsigned short* wq_h  = (unsigned short*)alloc(C_ * C_ * 2);
  unsigned short* wkv_h = (unsigned short*)alloc(2 * C_ * C_ * 2);
  unsigned short* wo_h  = (unsigned short*)alloc(C_ * C_ * 2);
  unsigned short* w1_h  = (unsigned short*)alloc(C_ * C_ * 2);
  unsigned short* w2_h  = (unsigned short*)alloc(C_ * C_ * 2);
  unsigned short* lnq   = (unsigned short*)alloc(actE * 2);       // also reused as qkvh
  unsigned short* lnkv  = (unsigned short*)alloc(2 * actE * 2);   // first half reused as lnx
  unsigned short* qh    = (unsigned short*)alloc(actE * 2);       // reused as h1
  unsigned short* kvh   = (unsigned short*)alloc(4 * actE * 2);   // reused as xbuf (f32, half size)
  float* qnorm = (float*)alloc(B_ * C_ * 4);
  float* knorm = (float*)alloc(2 * B_ * C_ * 4);
  unsigned short* attnw = (unsigned short*)alloc((size_t)B_ * NH_ * 2 * 64 * 64 * 2);
  unsigned short* qkvh = lnq;
  float* xbuf = (float*)kvh;
  unsigned short* lnx = lnkv;
  unsigned short* h1  = qh;

  // 1) weights -> bf16
  cvt_f32_bf16_kernel<<<(C_ * C_ + 255) / 256, 256, 0, stream>>>(Wq, wq_h, C_ * C_);
  cvt_f32_bf16_kernel<<<(2 * C_ * C_ + 255) / 256, 256, 0, stream>>>(Wkv, wkv_h, 2 * C_ * C_);
  cvt_f32_bf16_kernel<<<(C_ * C_ + 255) / 256, 256, 0, stream>>>(Wo, wo_h, C_ * C_);
  cvt_f32_bf16_kernel<<<(C_ * C_ + 255) / 256, 256, 0, stream>>>(W1, w1_h, C_ * C_);
  cvt_f32_bf16_kernel<<<(C_ * C_ + 255) / 256, 256, 0, stream>>>(W2, w2_h, C_ * C_);

  // 2) LayerNorms -> bf16 activations
  ln2d_kernel<<<B_ * HW_ / 256, 256, 0, stream>>>(img, q_ln_w, q_ln_b, lnq);
  ln2d_kernel<<<B_ * HW_ / 256, 256, 0, stream>>>(aux0, kv_ln_w, kv_ln_b, lnkv);
  ln2d_kernel<<<B_ * HW_ / 256, 256, 0, stream>>>(aux1, kv_ln_w, kv_ln_b, lnkv + actE);

  // 3) q / kv projections (WMMA)
  gemm_bf16_kernel<false, false, false, false, true>
      <<<dim3(HW_ / 128, C_ / 128, B_), 256, 0, stream>>>(
          wq_h, lnq, nullptr, qh, nullptr, nullptr, C_, C_, HW_);
  gemm_bf16_kernel<false, false, false, false, true>
      <<<dim3(HW_ / 128, 2 * C_ / 128, 2 * B_), 256, 0, stream>>>(
          wkv_h, lnkv, nullptr, kvh, nullptr, nullptr, 2 * C_, C_, HW_);

  // 4) row norms (q: all rows; k: channels 0..255 of each kv batch)
  rownorm_kernel<<<B_ * C_, 256, 0, stream>>>(qh, qnorm, C_, (size_t)C_ * HW_);
  rownorm_kernel<<<2 * B_ * C_, 256, 0, stream>>>(kvh, knorm, C_, (size_t)2 * C_ * HW_);

  // 5) cosine-attention logits + softmax (WMMA over K=4096)
  attn_logits_kernel<<<dim3(2, NH_, B_), 128, 0, stream>>>(qh, kvh, qnorm, knorm, attnw);

  // 6) AV + gate (WMMA, K=64)
  attn_av_kernel<<<dim3(HW_ / 64, NH_, B_), 128, 0, stream>>>(attnw, kvh, ascale, qkvh);

  // 7) x = Wo*qkv + bo + img   (fp32 out)
  gemm_bf16_kernel<true, true, false, true, false>
      <<<dim3(HW_ / 128, C_ / 128, B_), 256, 0, stream>>>(
          wo_h, qkvh, xbuf, nullptr, bo, img, C_, C_, HW_);

  // 8) MLP LN
  ln2d_kernel<<<B_ * HW_ / 256, 256, 0, stream>>>(xbuf, mlp_ln_w, mlp_ln_b, lnx);

  // 9) h1 = gelu(W1*lnx + b1) -> bf16
  gemm_bf16_kernel<true, false, true, false, true>
      <<<dim3(HW_ / 128, C_ / 128, B_), 256, 0, stream>>>(
          w1_h, lnx, nullptr, h1, b1, nullptr, C_, C_, HW_);

  // 10) out = W2*h1 + b2 + x  (fp32 final)
  gemm_bf16_kernel<true, true, false, true, false>
      <<<dim3(HW_ / 128, C_ / 128, B_), 256, 0, stream>>>(
          w2_h, h1, out, nullptr, b2, xbuf, C_, C_, HW_);

  (void)in_sizes; (void)n_in; (void)out_size; (void)ws_size;
}
